// MultiHeadAttention1_61907658605191
// MI455X (gfx1250) — matrix operand order repaired and checked by reading, not compiled, not run on hardware
//
#include <hip/hip_runtime.h>
#include <hip/hip_bf16.h>
#include <stdint.h>

typedef __attribute__((ext_vector_type(16))) __bf16 v16bf;
typedef __attribute__((ext_vector_type(8)))  __bf16 v8bf;
typedef __attribute__((ext_vector_type(8)))  float  v8f;
typedef __attribute__((ext_vector_type(4)))  unsigned int u32x4;
typedef __attribute__((ext_vector_type(4)))  int i32x4;
typedef __attribute__((ext_vector_type(8)))  int i32x8;

#define D_MODEL   1024
#define NUM_HEADS 16
#define HEAD_DIM  64
#define SEQ       2048
#define BATCH     2
#define ROWS      (BATCH * SEQ)   // 4096

// ---------------------------------------------------------------------------
// TDM: 2D bf16 tile load, global -> LDS (one descriptor, count=1).
//   group0: count=1 | lds_addr | global_addr[56:0] | type=2
//   group1: wg_mask=0 | data_size=1 (2B) | tensor dims | tile dims | stride0
//   groups 2/3: zero (tile_dim2..4 = 0 => unused; 2D tensor)
// Issue from ONE wave only (TDM ignores EXEC); track with TENSORcnt.
// ---------------------------------------------------------------------------
__device__ __forceinline__ void tdm_load_tile2d_bf16(
    const void* gptr, unsigned lds_off,
    unsigned tensor_d0, unsigned tensor_d1,      // elements
    unsigned tile_d0,   unsigned tile_d1,        // elements
    unsigned long long stride0_elems)
{
  const unsigned long long ga = (unsigned long long)(uintptr_t)gptr;
  u32x4 g0;
  g0[0] = 1u;                                        // count=1, user-mode
  g0[1] = lds_off;                                   // LDS byte address
  g0[2] = (unsigned)(ga & 0xffffffffull);            // global_addr[31:0]
  g0[3] = (unsigned)((ga >> 32) & 0x1ffffffull)      // global_addr[56:32]
        | (2u << 30);                                // type = 2 ("image")
  i32x8 g1;
  g1[0] = (int)(1u << 16);                           // data_size=1 (2 bytes)
  g1[1] = (int)((tensor_d0 & 0xffffu) << 16);        // tensor_dim0[15:0]
  g1[2] = (int)((tensor_d0 >> 16) | ((tensor_d1 & 0xffffu) << 16));
  g1[3] = (int)((tensor_d1 >> 16) | (tile_d0 << 16));// tile_dim0
  g1[4] = (int)(tile_d1 & 0xffffu);                  // tile_dim1 (tile_dim2=0)
  g1[5] = (int)(unsigned)(stride0_elems & 0xffffffffull);
  g1[6] = (int)(unsigned)((stride0_elems >> 32) & 0xffffull); // dim1_stride=0
  g1[7] = 0;
  const i32x4 z4 = {0, 0, 0, 0};
#if defined(__clang_major__) && (__clang_major__ >= 23)
  const i32x8 z8 = {0, 0, 0, 0, 0, 0, 0, 0};
  __builtin_amdgcn_tensor_load_to_lds(g0, g1, z4, z4, z8, 0);
#else
  __builtin_amdgcn_tensor_load_to_lds(g0, g1, z4, z4, 0);
#endif
}

// ---------------------------------------------------------------------------
// Generic GEMM:  C[M,N] = A[M,K](f32->bf16) * B[K,N](f32->bf16) + bias[N]
// Block: 256 threads = 8 waves (4x2 wave grid), each wave -> 32x32 output.
// BM=128, BN=64, BK=64.  LDS: A row-major, B transposed [n][k] so that the
// per-lane WMMA B-fragment (b[j] = B[hi*8+j][n], b[8+j] = B[16+hi*8+j][n]) is two contiguous reads.
// ---------------------------------------------------------------------------
#define BM  128
#define BN  64
#define BKT 64

__global__ __launch_bounds__(256)
void gemm_bias_wmma(const float* __restrict__ A, const float* __restrict__ Bm,
                    const float* __restrict__ bias, float* __restrict__ C,
                    int M, int N, int K)
{
  __shared__ __bf16 Alds[BM][BKT];   // 16 KB
  __shared__ __bf16 Blds[BN][BKT];   //  8 KB, transposed: [n][k]

  const int tid  = threadIdx.x;
  const int lane = tid & 31, wave = tid >> 5;
  const int hi   = lane >> 4, lm  = lane & 15;
  const int wm   = wave >> 1, wn  = wave & 1;
  const int rowBase = blockIdx.y * BM;
  const int colBase = blockIdx.x * BN;

  v8f acc[2][2] = {};

  const int ar = tid >> 1, ah = tid & 1;   // A staging: 2 threads / row of 64
  const int kr = tid >> 2, ns = tid & 3;   // B staging: 4 threads / row of 64

  for (int kk = 0; kk < K; kk += BKT) {
    if (kk + BKT < K) {   // gfx1250 global_prefetch_b8 for next K tile
      __builtin_prefetch(&A[(size_t)(rowBase + ar) * K + kk + BKT + ah * 32], 0, 0);
      __builtin_prefetch(&Bm[(size_t)(kk + BKT + kr) * N + colBase + ns * 16], 0, 0);
    }
    // stage A tile (128 x 64 f32 -> bf16)
    #pragma unroll
    for (int v = 0; v < 8; ++v) {
      const int c0 = ah * 32 + v * 4;
      float4 f = *(const float4*)&A[(size_t)(rowBase + ar) * K + kk + c0];
      Alds[ar][c0 + 0] = (__bf16)f.x; Alds[ar][c0 + 1] = (__bf16)f.y;
      Alds[ar][c0 + 2] = (__bf16)f.z; Alds[ar][c0 + 3] = (__bf16)f.w;
    }
    // stage B tile (64 x 64 f32 -> bf16), transposed into [n][k]
    #pragma unroll
    for (int v = 0; v < 4; ++v) {
      const int c0 = ns * 16 + v * 4;
      float4 f = *(const float4*)&Bm[(size_t)(kk + kr) * N + colBase + c0];
      Blds[c0 + 0][kr] = (__bf16)f.x; Blds[c0 + 1][kr] = (__bf16)f.y;
      Blds[c0 + 2][kr] = (__bf16)f.z; Blds[c0 + 3][kr] = (__bf16)f.w;
    }
    __syncthreads();

    #pragma unroll
    for (int kc = 0; kc < 2; ++kc) {     // two K=32 WMMA steps per staging
      v16bf afrag[2], bfrag[2];
      #pragma unroll
      for (int t = 0; t < 2; ++t) {
        // 16-bit A layout: lane half selects K octet (hi*8 and 16+hi*8)
        const v8bf lo = *(const v8bf*)&Alds[wm * 32 + t * 16 + lm][kc * 32 + hi * 8];
        const v8bf hv = *(const v8bf*)&Alds[wm * 32 + t * 16 + lm][kc * 32 + 16 + hi * 8];
        afrag[t] = __builtin_shufflevector(lo, hv, 0,1,2,3,4,5,6,7,8,9,10,11,12,13,14,15);
        // 16-bit B layout: same K octets as A (hi*8 and 16+hi*8) from Blds[n][k]
        const v8bf bl = *(const v8bf*)&Blds[wn * 32 + t * 16 + lm][kc * 32 + hi * 8];
        const v8bf bh = *(const v8bf*)&Blds[wn * 32 + t * 16 + lm][kc * 32 + 16 + hi * 8];
        bfrag[t] = __builtin_shufflevector(bl, bh, 0,1,2,3,4,5,6,7,8,9,10,11,12,13,14,15);
      }
      #pragma unroll
      for (int i = 0; i < 2; ++i)
        #pragma unroll
        for (int j = 0; j < 2; ++j)
          acc[i][j] = __builtin_amdgcn_wmma_f32_16x16x32_bf16(
              false, afrag[i], false, bfrag[j], (short)0, acc[i][j], false, false);
    }
    __syncthreads();
  }

  // epilogue: C/D layout -> row = hi*8 + r, col = lm (per 16x16 tile)
  #pragma unroll
  for (int i = 0; i < 2; ++i) {
    #pragma unroll
    for (int j = 0; j < 2; ++j) {
      const int col = colBase + wn * 32 + j * 16 + lm;
      const float bv = bias[col];
      #pragma unroll
      for (int r = 0; r < 8; ++r) {
        const int row = rowBase + wm * 32 + i * 16 + hi * 8 + r;
        C[(size_t)row * N + col] = acc[i][j][r] + bv;
      }
    }
  }
}

// ---------------------------------------------------------------------------
// RoPE + pack:  qkv f32 [B*S, 3D]  ->  Q,K (rope'd), V  bf16 [B,H,S,hd]
// ---------------------------------------------------------------------------
__global__ __launch_bounds__(256)
void rope_pack(const float* __restrict__ qkv, __bf16* __restrict__ Qb,
               __bf16* __restrict__ Kb, __bf16* __restrict__ Vb)
{
  const int idx = blockIdx.x * 256 + threadIdx.x;
  const int d = idx & 31;
  const int h = (idx >> 5) & 15;
  const int s = (idx >> 9) & 2047;
  const int b = idx >> 20;

  const size_t row  = (size_t)b * SEQ + s;
  const float* base = qkv + row * (3 * D_MODEL) + h * HEAD_DIM;
  const float q1 = base[d],                q2 = base[d + 32];
  const float k1 = base[D_MODEL + d],      k2 = base[D_MODEL + d + 32];
  const float v1 = base[2 * D_MODEL + d],  v2 = base[2 * D_MODEL + d + 32];

  // inv_freq = 10000^(-2d/64) = exp(-d * ln(10000)/32)
  const float freq = (float)s * __expf(-(float)d * 0.28782313662425572f);
  const float c = __cosf(freq), sn = __sinf(freq);

  const size_t ob = (((size_t)b * NUM_HEADS + h) * SEQ + s) * HEAD_DIM;
  Qb[ob + d]      = (__bf16)(q1 * c - q2 * sn);
  Qb[ob + d + 32] = (__bf16)(q2 * c + q1 * sn);
  Kb[ob + d]      = (__bf16)(k1 * c - k2 * sn);
  Kb[ob + d + 32] = (__bf16)(k2 * c + k1 * sn);
  Vb[ob + d]      = (__bf16)v1;
  Vb[ob + d + 32] = (__bf16)v2;
}

// ---------------------------------------------------------------------------
// Flash attention (causal), per (b,h): Q/K/V bf16 [S,64].
// Block = 128 threads = 4 waves; q-tile = 64 rows (16 per wave); KV chunk = 64.
// Q/K tiles staged by the Tensor Data Mover (wave 0 issues, TENSORcnt wait,
// then workgroup barrier publishes LDS).  V staged transposed by VALU (TDM
// cannot transpose).  P converts C-layout -> A-layout through per-wave LDS.
// ---------------------------------------------------------------------------
__global__ __launch_bounds__(128)
void attn_fwd(const __bf16* __restrict__ Q, const __bf16* __restrict__ Kk,
              const __bf16* __restrict__ V, float* __restrict__ O)
{
  __shared__ __bf16 Qs[64][64];       // 8 KB  (TDM destination)
  __shared__ __bf16 Ks[64][64];       // 8 KB  (TDM destination)
  __shared__ __bf16 Vt[64][64];       // 8 KB, [d][kc]
  __shared__ __bf16 Ps[4][16][64];    // 8 KB, per-wave P tile

  const int tid  = threadIdx.x;
  const int lane = tid & 31, wave = tid >> 5;
  const int hi   = lane >> 4, lm  = lane & 15;
  const int qtile = blockIdx.x;            // 0..31
  const int bh    = blockIdx.y;            // 0..31
  const int qbase = qtile * 64;

  const __bf16* Qp  = Q  + ((size_t)bh * SEQ + qbase) * HEAD_DIM;
  const __bf16* Kp0 = Kk + (size_t)bh * SEQ * HEAD_DIM;
  const __bf16* Vp0 = V  + (size_t)bh * SEQ * HEAD_DIM;

  const unsigned qs_lds = (unsigned)(uintptr_t)&Qs[0][0];
  const unsigned ks_lds = (unsigned)(uintptr_t)&Ks[0][0];

  // Q tile: one TDM descriptor (64x64 bf16, row stride 64) issued by wave 0.
  if (wave == 0)
    tdm_load_tile2d_bf16(Qp, qs_lds, HEAD_DIM, SEQ, 64, 64, HEAD_DIM);

  v8f o[4] = {};
  float mrow[8], lrow[8];
  #pragma unroll
  for (int r = 0; r < 8; ++r) { mrow[r] = -3.0e38f; lrow[r] = 0.f; }

  const int nch = qtile + 1;               // causal: only chunks with k <= q
  for (int ch = 0; ch < nch; ++ch) {
    const int kvbase = ch * 64;
    __syncthreads();                       // previous iteration's LDS reads done
    // K chunk via TDM (wave 0), overlapped with manual transposed V staging.
    if (wave == 0)
      tdm_load_tile2d_bf16(Kp0 + (size_t)kvbase * HEAD_DIM, ks_lds,
                           HEAD_DIM, SEQ, 64, 64, HEAD_DIM);
    {
      const int r = tid >> 1, h2 = tid & 1;
      const __bf16* Vp = Vp0 + (size_t)(kvbase + r) * 64;
      #pragma unroll
      for (int v = 0; v < 4; ++v) {
        v8bf tv = *(const v8bf*)&Vp[h2 * 32 + v * 8];
        #pragma unroll
        for (int e = 0; e < 8; ++e)
          Vt[h2 * 32 + v * 8 + e][r] = tv[e];
      }
    }
    if (wave == 0)
      __builtin_amdgcn_s_wait_tensorcnt(0);  // Q (iter 0) + K tile landed
    __syncthreads();

    // ---- S = Q * K^T : wave computes 16 q rows x 64 k cols
    v16bf aq[2];
    #pragma unroll
    for (int kc = 0; kc < 2; ++kc) {
      const v8bf lo = *(const v8bf*)&Qs[wave * 16 + lm][kc * 32 + hi * 8];
      const v8bf hv = *(const v8bf*)&Qs[wave * 16 + lm][kc * 32 + 16 + hi * 8];
      aq[kc] = __builtin_shufflevector(lo, hv, 0,1,2,3,4,5,6,7,8,9,10,11,12,13,14,15);
    }
    v8f sacc[4] = {};
    #pragma unroll
    for (int nt = 0; nt < 4; ++nt)
      #pragma unroll
      for (int kc = 0; kc < 2; ++kc) {
        const v8bf bl = *(const v8bf*)&Ks[nt * 16 + lm][kc * 32 + hi * 8];
        const v8bf bh = *(const v8bf*)&Ks[nt * 16 + lm][kc * 32 + 16 + hi * 8];
        v16bf bk = __builtin_shufflevector(bl, bh, 0,1,2,3,4,5,6,7,8,9,10,11,12,13,14,15);
        sacc[nt] = __builtin_amdgcn_wmma_f32_16x16x32_bf16(
            false, aq[kc], false, bk, (short)0, sacc[nt], false, false);
      }

    // ---- scale + causal mask
    float sv[4][8];
    #pragma unroll
    for (int nt = 0; nt < 4; ++nt) {
      const int kg = kvbase + nt * 16 + lm;
      #pragma unroll
      for (int r = 0; r < 8; ++r) {
        const int qg = qbase + wave * 16 + hi * 8 + r;
        const float s = sacc[nt][r] * 0.125f;   // 1/sqrt(64)
        sv[nt][r] = (kg > qg) ? -3.0e38f : s;
      }
    }

    // ---- online softmax (rows live across the 16 lanes of each half-wave)
    #pragma unroll
    for (int r = 0; r < 8; ++r) {
      float rm = fmaxf(fmaxf(sv[0][r], sv[1][r]), fmaxf(sv[2][r], sv[3][r]));
      #pragma unroll
      for (int m = 1; m < 16; m <<= 1) rm = fmaxf(rm, __shfl_xor(rm, m, 32));
      const float mnew  = fmaxf(mrow[r], rm);
      const float alpha = __expf(mrow[r] - mnew);
      mrow[r] = mnew;
      float rs = 0.f;
      #pragma unroll
      for (int nt = 0; nt < 4; ++nt) {
        const float p = __expf(sv[nt][r] - mnew);
        Ps[wave][hi * 8 + r][nt * 16 + lm] = (__bf16)p;   // C-layout -> LDS
        rs += p;
      }
      #pragma unroll
      for (int m = 1; m < 16; m <<= 1) rs += __shfl_xor(rs, m, 32);
      lrow[r] = lrow[r] * alpha + rs;
      #pragma unroll
      for (int dt = 0; dt < 4; ++dt) o[dt][r] *= alpha;
    }

    // cross-lane LDS hazard (P write -> A-layout read) is invisible to the
    // compiler's per-thread analysis; enforce DS completion explicitly.
    asm volatile("s_wait_dscnt 0" ::: "memory");

    // ---- O += P * V
    v16bf ap[2];
    #pragma unroll
    for (int pc = 0; pc < 2; ++pc) {
      const v8bf lo = *(const v8bf*)&Ps[wave][lm][pc * 32 + hi * 8];
      const v8bf hv = *(const v8bf*)&Ps[wave][lm][pc * 32 + 16 + hi * 8];
      ap[pc] = __builtin_shufflevector(lo, hv, 0,1,2,3,4,5,6,7,8,9,10,11,12,13,14,15);
    }
    #pragma unroll
    for (int dt = 0; dt < 4; ++dt)
      #pragma unroll
      for (int pc = 0; pc < 2; ++pc) {
        const v8bf bl = *(const v8bf*)&Vt[dt * 16 + lm][pc * 32 + hi * 8];
        const v8bf bh = *(const v8bf*)&Vt[dt * 16 + lm][pc * 32 + 16 + hi * 8];
        v16bf bv = __builtin_shufflevector(bl, bh, 0,1,2,3,4,5,6,7,8,9,10,11,12,13,14,15);
        o[dt] = __builtin_amdgcn_wmma_f32_16x16x32_bf16(
            false, ap[pc], false, bv, (short)0, o[dt], false, false);
      }
  }

  // ---- normalize + scatter to [B, S, D] f32 (feeds out-projection GEMM)
  const int b = bh >> 4, h = bh & 15;
  #pragma unroll
  for (int r = 0; r < 8; ++r) {
    const float inv = 1.0f / lrow[r];
    const int qg = qbase + wave * 16 + hi * 8 + r;
    float* Orow = O + ((size_t)b * SEQ + qg) * D_MODEL + h * HEAD_DIM;
    #pragma unroll
    for (int dt = 0; dt < 4; ++dt)
      Orow[dt * 16 + lm] = o[dt][r] * inv;
  }
}

// ---------------------------------------------------------------------------
extern "C" void kernel_launch(void* const* d_in, const int* in_sizes, int n_in,
                              void* d_out, int out_size, void* d_ws, size_t ws_size,
                              hipStream_t stream)
{
  (void)in_sizes; (void)n_in; (void)out_size; (void)ws_size;
  const float* x    = (const float*)d_in[0];   // [2,2048,1024]
  const float* Wqkv = (const float*)d_in[1];   // [1024,3072]
  const float* bqkv = (const float*)d_in[2];   // [3072]
  const float* Wout = (const float*)d_in[3];   // [1024,1024]
  const float* bout = (const float*)d_in[4];   // [1024]
  float* out = (float*)d_out;                  // [2,2048,1024]

  char* w = (char*)d_ws;
  float* qkv = (float*)w;                                   // 50.3 MB
  size_t off = (size_t)ROWS * 3 * D_MODEL * sizeof(float);
  __bf16* Qb = (__bf16*)(w + off); off += (size_t)ROWS * D_MODEL * sizeof(__bf16);
  __bf16* Kb = (__bf16*)(w + off); off += (size_t)ROWS * D_MODEL * sizeof(__bf16);
  __bf16* Vb = (__bf16*)(w + off); off += (size_t)ROWS * D_MODEL * sizeof(__bf16);
  float* attnO = (float*)(w + off);                         // 16.8 MB

  // 1) qkv = x @ W_qkv + b_qkv
  gemm_bias_wmma<<<dim3(3 * D_MODEL / BN, ROWS / BM), 256, 0, stream>>>(
      x, Wqkv, bqkv, qkv, ROWS, 3 * D_MODEL, D_MODEL);
  // 2) RoPE + head-major bf16 pack
  rope_pack<<<dim3((ROWS * NUM_HEADS * 32) / 256), 256, 0, stream>>>(qkv, Qb, Kb, Vb);
  // 3) causal flash attention (TDM-staged Q/K)
  attn_fwd<<<dim3(SEQ / 64, BATCH * NUM_HEADS), 128, 0, stream>>>(Qb, Kb, Vb, attnO);
  // 4) out = attnO @ W_out + b_out
  gemm_bias_wmma<<<dim3(D_MODEL / BN, ROWS / BM), 256, 0, stream>>>(
      attnO, Wout, bout, out, ROWS, D_MODEL, D_MODEL);
}
